// VoxelFeatureExtractor_2147483648163
// MI455X (gfx1250) — matrix operand order repaired and checked by reading, not compiled, not run on hardware
//
#include <hip/hip_runtime.h>

#define N_VOX   40000
#define T_PTS   32
#define C_IN    8
#define P_SLOTS 16
#define BEV_H   496
#define BEV_W   432
#define N_CELL  (BEV_H * BEV_W)
#define EPSV    1e-5f

typedef __attribute__((ext_vector_type(16))) _Float16 v16h;
typedef __attribute__((ext_vector_type(8)))  _Float16 v8h;
typedef __attribute__((ext_vector_type(8)))  float    v8f;
typedef __attribute__((ext_vector_type(4)))  unsigned v4u;
typedef __attribute__((ext_vector_type(8)))  unsigned v8u;

// Fast Mish: mish(y) = y*tanh(softplus(y)); with t = 1+e^y,
// tanh(ln t) = (t^2-1)/(t^2+1). One v_exp_f32 + one v_rcp_f32.
__device__ __forceinline__ float mishf(float y) {
    float e  = __expf(y);
    float t  = 1.f + e;
    float t2 = t * t;
    float r  = (t2 - 1.f) * __builtin_amdgcn_rcpf(t2 + 1.f);
    return (y > 15.f) ? y : y * r;
}
__device__ __forceinline__ float bnf(float y, float g, float b, float m, float v) {
    return (y - m) * __builtin_amdgcn_rsqf(v + EPSV) * g + b;
}
// Build a 16xf16 WMMA fragment from two 16-byte LDS chunks (ds_load_b128 x2).
__device__ __forceinline__ v16h ld16h(const _Float16* p0, const _Float16* p1) {
    v8h a = *(const v8h*)p0;
    v8h b = *(const v8h*)p1;
    return __builtin_shufflevector(a, b, 0,1,2,3,4,5,6,7,8,9,10,11,12,13,14,15);
}

// ---------------------------------------------------------------------------
// Tensor Data Mover: 1-D tile copy global -> LDS (ndwords 4-byte elements).
// D# per CDNA5 ISA 8.3/8.4: group0 = {count=1, lds_addr, global_addr, type=2};
// group1 = {data_size=2(4B), tensor_dim0 = tile_dim0 = stride = ndwords}.
// Issue from ONE wave only (EXEC ignored; every wave reaching it issues a DMA).
// ---------------------------------------------------------------------------
__device__ __forceinline__ void tdm_load_1d(const void* gsrc, void* lds_dst,
                                            unsigned ndwords) {
    unsigned long long ga = (unsigned long long)(uintptr_t)gsrc;
    v4u g0;
    g0[0] = 1u;                                            // count=1 (valid user D#)
    g0[1] = (unsigned)(uintptr_t)lds_dst;                  // LDS byte address
    g0[2] = (unsigned)ga;                                  // global_addr[31:0]
    g0[3] = ((unsigned)(ga >> 32) & 0x01FFFFFFu) | (2u << 30); // addr[56:32] | type=2
    v8u g1;
    g1[0] = 2u << 16;                                      // data_size=2 -> 4 bytes
    g1[1] = (ndwords & 0xFFFFu) << 16;                     // tensor_dim0[15:0]
    g1[2] = (ndwords >> 16) | (1u << 16);                  // tensor_dim0[31:16], tensor_dim1=1
    g1[3] = (ndwords & 0xFFFFu) << 16;                     // tile_dim0
    g1[4] = 0u;                                            // tile_dim1=0, tile_dim2=0
    g1[5] = ndwords;                                       // tensor_dim0_stride[31:0]
    g1[6] = 0u;
    g1[7] = 0u;
    asm volatile("tensor_load_to_lds %0, %1" :: "s"(g0), "s"(g1) : "memory");
}
__device__ __forceinline__ void tdm_wait0() {
    asm volatile("s_wait_tensorcnt 0x0" ::: "memory");
}

// ---------------------------------------------------------------------------
// Init: zero output bevmap, reset grouping state
// ---------------------------------------------------------------------------
__global__ void zero_init_kernel(float* out, long out_n, int* cellmap,
                                 int* slotcnt, int* nrows) {
    long i = (long)blockIdx.x * blockDim.x + threadIdx.x;
    long stride = (long)gridDim.x * blockDim.x;
    for (long k = i; k < out_n; k += stride) out[k] = 0.f;
    for (long k = i; k < N_CELL; k += stride) cellmap[k] = -1;
    for (long k = i; k < N_VOX; k += stride) slotcnt[k] = 0;
    if (i == 0) *nrows = 0;
}

// One-time f32 -> f16 weight conversion (so blocks can TDM-copy them raw).
__global__ void prep_weights_kernel(const float* __restrict__ W3,
                                    const float* __restrict__ W4,
                                    const float* __restrict__ Wb3,
                                    _Float16* __restrict__ w3h,
                                    _Float16* __restrict__ w4h,
                                    _Float16* __restrict__ wb3h) {
    int i = blockIdx.x * blockDim.x + threadIdx.x;
    if (i < 4096) {
        w3h[i] = (_Float16)W3[i];
        w4h[i] = (_Float16)W4[i];
    }
    if (i < 16384) wb3h[i] = (_Float16)Wb3[i];
}

// ---------------------------------------------------------------------------
// voxel -> BEV cell grouping (replaces unique/argsort with atomics)
// ---------------------------------------------------------------------------
__global__ void cell_flag_kernel(const int* __restrict__ coors, int* cellmap) {
    int v = blockIdx.x * blockDim.x + threadIdx.x;
    if (v >= N_VOX) return;
    int cell = coors[2 * v] * BEV_W + coors[2 * v + 1];
    atomicCAS(&cellmap[cell], -1, -2);
}

__global__ void cell_assign_kernel(int* cellmap, int* row_cell, int* nrows) {
    int c = blockIdx.x * blockDim.x + threadIdx.x;
    if (c >= N_CELL) return;
    if (cellmap[c] == -2) {
        int r = atomicAdd(nrows, 1);
        cellmap[c] = r;
        row_cell[r] = c;
    }
}

__global__ void slot_assign_kernel(const int* __restrict__ coors,
                                   const int* __restrict__ cellmap,
                                   int* slotcnt, int* rowvox) {
    int v = blockIdx.x * blockDim.x + threadIdx.x;
    if (v >= N_VOX) return;
    int cell = coors[2 * v] * BEV_W + coors[2 * v + 1];
    int row  = cellmap[cell];
    int slot = atomicAdd(&slotcnt[row], 1);
    if (slot < P_SLOTS) rowvox[row * P_SLOTS + slot] = v;
}

// ---------------------------------------------------------------------------
// VFE: one block per voxel. vfe1/vfe2 in f32 VALU (overlapped with TDM weight
// DMA), vfe3/vfe4 via f16 WMMA.
// ---------------------------------------------------------------------------
__global__ __launch_bounds__(256) void vfe_kernel(
    const float* __restrict__ feat, const int* __restrict__ numvox,
    const float* __restrict__ W1, const float* __restrict__ bn1,
    const float* __restrict__ W2, const float* __restrict__ bn2,
    const _Float16* __restrict__ w3h_g, const float* __restrict__ bn3,
    const _Float16* __restrict__ w4h_g, const float* __restrict__ bn4,
    float* __restrict__ voxelwise)
{
    __shared__ float    sf[T_PTS][C_IN];
    __shared__ float    sx1[T_PTS][16];
    __shared__ float    sx2f[T_PTS][64];
    __shared__ __align__(16) _Float16 sx2h[T_PTS][64];
    __shared__ __align__(16) _Float16 sx3h[T_PTS][64];
    __shared__ __align__(16) _Float16 sw3[64][64];
    __shared__ __align__(16) _Float16 sw4[64][64];
    __shared__ float    sbn3[4][64];
    __shared__ float    sbn4[4][64];
    __shared__ float    scm[32];
    __shared__ float    spart[8][32];

    int v   = blockIdx.x;
    int tid = threadIdx.x;
    int nv  = numvox[v];

    // Kick off async weight DMA (one wave issues; overlaps vfe1/vfe2 below)
    if (tid == 0) {
        tdm_load_1d(w3h_g, &sw3[0][0], 2048);   // 8 KB
        tdm_load_1d(w4h_g, &sw4[0][0], 2048);   // 8 KB
    }
    if (tid < 64) {
        for (int j = 0; j < 4; j++) {
            sbn3[j][tid] = bn3[j * 64 + tid];
            sbn4[j][tid] = bn4[j * 64 + tid];
        }
    }
    sf[tid >> 3][tid & 7] = feat[v * (T_PTS * C_IN) + tid];
    __syncthreads();

    // ---- vfe1: Linear(8->8)+BN+Mish, one output per thread ----
    {
        int t = tid >> 3, o = tid & 7;
        float acc = 0.f;
        for (int i = 0; i < C_IN; i++) acc += sf[t][i] * W1[o * C_IN + i];
        float y = bnf(acc, bn1[o], bn1[8 + o], bn1[16 + o], bn1[24 + o]);
        sx1[t][o] = mishf(y);
    }
    __syncthreads();
    if (tid < 8) {
        float m = -1e30f;
        for (int t = 0; t < T_PTS; t++) m = fmaxf(m, sx1[t][tid]);
        scm[tid] = m;
    }
    __syncthreads();
    { int t = tid >> 3, o = tid & 7; sx1[t][8 + o] = scm[o]; }
    __syncthreads();

    // ---- vfe2: Linear(16->32)+BN+Mish, 4 outputs per thread ----
    {
        int t = tid >> 3, ob = (tid & 7) * 4;
        for (int j = 0; j < 4; j++) {
            int o = ob + j;
            float acc = 0.f;
            for (int i = 0; i < 16; i++) acc += sx1[t][i] * W2[o * 16 + i];
            float y = bnf(acc, bn2[o], bn2[32 + o], bn2[64 + o], bn2[96 + o]);
            sx2f[t][o] = mishf(y);
        }
    }
    __syncthreads();
    if (tid < 32) {
        float m = -1e30f;
        for (int t = 0; t < T_PTS; t++) m = fmaxf(m, sx2f[t][tid]);
        scm[tid] = m;
    }
    __syncthreads();
    {   // x2 = concat(p2, colmax) * mask   -> f32 (residual) + f16 (WMMA A)
        int t = tid >> 3, cb = (tid & 7) * 8;
        for (int j = 0; j < 8; j++) {
            int c = cb + j;
            float val = (c < 32) ? sx2f[t][c] : scm[c - 32];
            if (t >= nv) val = 0.f;
            sx2f[t][c] = val;
            sx2h[t][c] = (_Float16)val;
        }
    }
    if (tid == 0) tdm_wait0();     // weights must have landed in LDS
    __syncthreads();

    // ---- vfe3: [32,64] x [64,64]^T via WMMA, 2x4 tiles over 8 waves ----
    int wave = tid >> 5, lane = tid & 31;
    int mi = wave >> 2, ni = wave & 3;
    int l15 = lane & 15, hi = lane >> 4;
    int ncol = ni * 16 + l15;
    {
        v8f acc = {};
        for (int k0 = 0; k0 < 64; k0 += 32) {
            int M  = mi * 16 + l15;
            int ab = k0 + hi * 8;
            v16h a = ld16h(&sx2h[M][ab], &sx2h[M][ab + 16]);
            int bb = k0 + hi * 8;
            v16h b = ld16h(&sw3[ncol][bb], &sw3[ncol][bb + 16]);
            acc = __builtin_amdgcn_wmma_f32_16x16x32_f16(false, a, false, b,
                                                         (short)0, acc, false, false);
        }
        float g = sbn3[0][ncol], be = sbn3[1][ncol], mm = sbn3[2][ncol], vv = sbn3[3][ncol];
        float s = __builtin_amdgcn_rsqf(vv + EPSV) * g;
        for (int r = 0; r < 8; r++) {
            int M = mi * 16 + r + hi * 8;
            float y = (acc[r] - mm) * s + be;
            sx3h[M][ncol] = (_Float16)mishf(y);
        }
    }
    __syncthreads();

    // ---- vfe4: WMMA + BN + Mish*mask + x2, then column max over T ----
    {
        v8f acc = {};
        for (int k0 = 0; k0 < 64; k0 += 32) {
            int M  = mi * 16 + l15;
            int ab = k0 + hi * 8;
            v16h a = ld16h(&sx3h[M][ab], &sx3h[M][ab + 16]);
            int bb = k0 + hi * 8;
            v16h b = ld16h(&sw4[ncol][bb], &sw4[ncol][bb + 16]);
            acc = __builtin_amdgcn_wmma_f32_16x16x32_f16(false, a, false, b,
                                                         (short)0, acc, false, false);
        }
        float g = sbn4[0][ncol], be = sbn4[1][ncol], mm = sbn4[2][ncol], vv = sbn4[3][ncol];
        float s = __builtin_amdgcn_rsqf(vv + EPSV) * g;
        float pm = -1e30f;
        for (int r = 0; r < 8; r++) {
            int M = mi * 16 + r + hi * 8;
            float y  = (acc[r] - mm) * s + be;
            float x4 = mishf(y);
            if (M >= nv) x4 = 0.f;
            x4 += sx2f[M][ncol];
            pm = fmaxf(pm, x4);
        }
        spart[wave][lane] = pm;
    }
    __syncthreads();
    if (tid < 64) {
        int nii = tid >> 4, cc = tid & 15;
        float m = fmaxf(fmaxf(spart[nii][cc], spart[nii][cc + 16]),
                        fmaxf(spart[4 + nii][cc], spart[4 + nii][cc + 16]));
        voxelwise[v * 64 + tid] = m;
    }
}

// ---------------------------------------------------------------------------
// bfe1 + bfe2: tiny grouped GEMMs (M=4,K=16) -> f32 VALU. 4 rows per block.
// ---------------------------------------------------------------------------
__global__ __launch_bounds__(256) void bfe12_kernel(
    const float* __restrict__ voxelwise, const int* __restrict__ rowvox,
    const int* __restrict__ slotcnt, const int* __restrict__ nrows_ptr,
    const float* __restrict__ W1g, const float* __restrict__ bn1g,
    const float* __restrict__ W2g, const float* __restrict__ bn2g,
    float* __restrict__ x17)
{
    __shared__ float sw1[16 * 16 * 16];
    __shared__ float sbnA[16 * 4 * 16];
    __shared__ float sw2[16 * 2 * 16];
    __shared__ float sbnB[16 * 4 * 2];
    __shared__ float sx0[4][64][P_SLOTS];
    __shared__ float sy[4][64][P_SLOTS];
    __shared__ int   svid[4][P_SLOTS];
    __shared__ int   snv[4];

    int tid   = threadIdx.x;
    int nrows = *nrows_ptr;
    int row0  = blockIdx.x * 4;
    if (row0 >= nrows) return;

    for (int j = tid; j < 4096; j += 256) sw1[j]  = W1g[j];
    for (int j = tid; j < 1024; j += 256) sbnA[j] = bn1g[j];
    for (int j = tid; j < 512;  j += 256) sw2[j]  = W2g[j];
    for (int j = tid; j < 128;  j += 256) sbnB[j] = bn2g[j];

    int r    = tid >> 6;     // local row 0..3
    int tr   = tid & 63;
    int grow = row0 + r;
    if (tr == 0) snv[r] = (grow < nrows) ? min(slotcnt[grow], P_SLOTS) : 0;
    __syncthreads();
    int numv = snv[r];
    if (tr < P_SLOTS) svid[r][tr] = (tr < numv) ? rowvox[grow * P_SLOTS + tr] : -1;
    __syncthreads();
    for (int p = 0; p < P_SLOTS; p++) {
        int vid = svid[r][p];
        sx0[r][tr][p] = (vid >= 0) ? voxelwise[vid * 64 + tr] : 0.f;
    }
    __syncthreads();

    // bfe1: y[g,i,q] = sum_p x0[g,i,p] * W1g[g,q,p]; BN(g,q); Mish; mask q<numv
    for (int jj = 0; jj < 16; jj++) {
        int o = tr + 64 * jj;     // 0..1023 = (g*4+i)*16 + q
        int c = o >> 4;
        int q = o & 15;
        int g = c >> 2;
        float acc = 0.f;
        const float* wrow = &sw1[(g * 16 + q) * 16];
        for (int p = 0; p < P_SLOTS; p++) acc += sx0[r][c][p] * wrow[p];
        float yy = bnf(acc, sbnA[g * 64 + q], sbnA[g * 64 + 16 + q],
                       sbnA[g * 64 + 32 + q], sbnA[g * 64 + 48 + q]);
        yy = mishf(yy);
        if (q >= numv) yy = 0.f;
        sy[r][c][q] = yy;
    }
    __syncthreads();

    // bfe2: z[g,i,c2] = sum_q y[g,i,q] * W2g[g,c2,q]; BN(g,c2); Mish; x17 layout
    for (int jj = 0; jj < 2; jj++) {
        int o  = tr + 64 * jj;    // 0..127 = (g*4+i)*2 + c2
        int c2 = o & 1;
        int gi = o >> 1;
        int g  = gi >> 2;
        int i  = gi & 3;
        float acc = 0.f;
        const float* wrow = &sw2[(g * 2 + c2) * 16];
        for (int q = 0; q < P_SLOTS; q++) acc += sy[r][gi][q] * wrow[q];
        float zz = bnf(acc, sbnB[g * 8 + c2], sbnB[g * 8 + 2 + c2],
                       sbnB[g * 8 + 4 + c2], sbnB[g * 8 + 6 + c2]);
        zz = mishf(zz);
        int oidx = ((i * 16) + g) * 2 + c2;   // transpose(0,2,1,3) flatten
        if (grow < nrows) x17[grow * 128 + oidx] = zz;
    }
}

// ---------------------------------------------------------------------------
// bfe3 applied twice: [32,128] x [128,128]^T via WMMA (weights TDM'd to LDS),
// then scatter to bevmap.
// ---------------------------------------------------------------------------
__global__ __launch_bounds__(256) void bfe3_kernel(
    const float* __restrict__ x17, const int* __restrict__ row_cell,
    const int* __restrict__ nrows_ptr,
    const _Float16* __restrict__ wb3h_g, const float* __restrict__ bn3,
    float* __restrict__ out)
{
    __shared__ __align__(16) _Float16 sA[32][128];
    __shared__ __align__(16) _Float16 sW[128][128];
    __shared__ __align__(16) _Float16 sB[32][128];
    __shared__ float    sbn[4][128];

    int tid   = threadIdx.x;
    int nrows = *nrows_ptr;
    int row0  = blockIdx.x * 32;
    if (row0 >= nrows) return;

    if (tid == 0) tdm_load_1d(wb3h_g, &sW[0][0], 8192);   // 32 KB weight DMA
    for (int j = tid; j < 512; j += 256) ((float*)sbn)[j] = bn3[j];
    for (int j = tid; j < 32 * 128; j += 256) {
        int rr = j >> 7;
        ((_Float16*)sA)[j] = (row0 + rr < nrows) ? (_Float16)x17[(row0 + rr) * 128 + (j & 127)]
                                                 : (_Float16)0.f;
    }
    if (tid == 0) tdm_wait0();
    __syncthreads();

    int wave = tid >> 5, lane = tid & 31;
    int l15 = lane & 15, hi = lane >> 4;
    int ni = wave;                    // 8 waves cover N; each wave does M tiles 0 and 1
    int ncol = ni * 16 + l15;
    float g = sbn[0][ncol], be = sbn[1][ncol], mm = sbn[2][ncol], vv = sbn[3][ncol];
    float s = __builtin_amdgcn_rsqf(vv + EPSV) * g;

    // ---- GEMM 1 ----
    v8f acc0 = {}, acc1 = {};
    for (int k0 = 0; k0 < 128; k0 += 32) {
        int bb = k0 + hi * 8;
        v16h b  = ld16h(&sW[ncol][bb], &sW[ncol][bb + 16]);
        int ab = k0 + hi * 8;
        v16h a0 = ld16h(&sA[l15][ab],      &sA[l15][ab + 16]);
        v16h a1 = ld16h(&sA[16 + l15][ab], &sA[16 + l15][ab + 16]);
        acc0 = __builtin_amdgcn_wmma_f32_16x16x32_f16(false, a0, false, b, (short)0, acc0, false, false);
        acc1 = __builtin_amdgcn_wmma_f32_16x16x32_f16(false, a1, false, b, (short)0, acc1, false, false);
    }
    for (int r = 0; r < 8; r++) {
        int M0 = r + hi * 8;
        float y0 = (acc0[r] - mm) * s + be;
        sB[M0][ncol]      = (_Float16)mishf(y0);
        float y1 = (acc1[r] - mm) * s + be;
        sB[16 + M0][ncol] = (_Float16)mishf(y1);
    }
    __syncthreads();

    // ---- GEMM 2 (same weights/BN: bfe3 applied twice) ----
    v8f acc2 = {}, acc3 = {};
    for (int k0 = 0; k0 < 128; k0 += 32) {
        int bb = k0 + hi * 8;
        v16h b  = ld16h(&sW[ncol][bb], &sW[ncol][bb + 16]);
        int ab = k0 + hi * 8;
        v16h a0 = ld16h(&sB[l15][ab],      &sB[l15][ab + 16]);
        v16h a1 = ld16h(&sB[16 + l15][ab], &sB[16 + l15][ab + 16]);
        acc2 = __builtin_amdgcn_wmma_f32_16x16x32_f16(false, a0, false, b, (short)0, acc2, false, false);
        acc3 = __builtin_amdgcn_wmma_f32_16x16x32_f16(false, a1, false, b, (short)0, acc3, false, false);
    }
    // BN + Mish + scatter to out[c][x][y]  (bevmap.transpose(2,1,0))
    for (int r = 0; r < 8; r++) {
        int M = r + hi * 8;
        int grow = row0 + M;
        if (grow < nrows) {
            float y = (acc2[r] - mm) * s + be;
            int cell = row_cell[grow];
            int cy = cell / BEV_W, cx = cell % BEV_W;
            out[(long)ncol * (BEV_W * BEV_H) + cx * BEV_H + cy] = mishf(y);
        }
    }
    for (int r = 0; r < 8; r++) {
        int M = 16 + r + hi * 8;
        int grow = row0 + M;
        if (grow < nrows) {
            float y = (acc3[r] - mm) * s + be;
            int cell = row_cell[grow];
            int cy = cell / BEV_W, cx = cell % BEV_W;
            out[(long)ncol * (BEV_W * BEV_H) + cx * BEV_H + cy] = mishf(y);
        }
    }
}

// ---------------------------------------------------------------------------
extern "C" void kernel_launch(void* const* d_in, const int* in_sizes, int n_in,
                              void* d_out, int out_size, void* d_ws, size_t ws_size,
                              hipStream_t stream)
{
    (void)in_sizes; (void)n_in; (void)ws_size;
    const float* features = (const float*)d_in[0];
    const int*   coors    = (const int*)d_in[1];
    const int*   numvox   = (const int*)d_in[2];
    const float* vfe1_W   = (const float*)d_in[3];
    const float* vfe1_bn  = (const float*)d_in[4];
    const float* vfe2_W   = (const float*)d_in[5];
    const float* vfe2_bn  = (const float*)d_in[6];
    const float* vfe3_W   = (const float*)d_in[7];
    const float* vfe3_bn  = (const float*)d_in[8];
    const float* vfe4_W   = (const float*)d_in[9];
    const float* vfe4_bn  = (const float*)d_in[10];
    const float* bfe1_W   = (const float*)d_in[11];
    const float* bfe1_bn  = (const float*)d_in[12];
    const float* bfe2_W   = (const float*)d_in[13];
    const float* bfe2_bn  = (const float*)d_in[14];
    const float* bfe3_W   = (const float*)d_in[15];
    const float* bfe3_bn  = (const float*)d_in[16];

    float* out = (float*)d_out;

    // workspace layout
    float* voxelwise = (float*)d_ws;                         // N*64
    float* x17       = voxelwise + (size_t)N_VOX * 64;       // N*128
    int*   cellmap   = (int*)(x17 + (size_t)N_VOX * 128);    // N_CELL
    int*   row_cell  = cellmap + N_CELL;                     // N
    int*   rowvox    = row_cell + N_VOX;                     // N*P
    int*   slotcnt   = rowvox + N_VOX * P_SLOTS;             // N
    int*   nrows     = slotcnt + N_VOX;                      // 1
    uintptr_t p = (uintptr_t)(nrows + 1);
    p = (p + 255) & ~(uintptr_t)255;                         // 256B align for TDM
    _Float16* w3h  = (_Float16*)p;                           // 4096
    _Float16* w4h  = w3h + 4096;                             // 4096
    _Float16* wb3h = w4h + 4096;                             // 16384

    zero_init_kernel<<<4096, 256, 0, stream>>>(out, (long)out_size, cellmap, slotcnt, nrows);
    prep_weights_kernel<<<(16384 + 255) / 256, 256, 0, stream>>>(vfe3_W, vfe4_W, bfe3_W,
                                                                 w3h, w4h, wb3h);
    cell_flag_kernel<<<(N_VOX + 255) / 256, 256, 0, stream>>>(coors, cellmap);
    cell_assign_kernel<<<(N_CELL + 255) / 256, 256, 0, stream>>>(cellmap, row_cell, nrows);
    slot_assign_kernel<<<(N_VOX + 255) / 256, 256, 0, stream>>>(coors, cellmap, slotcnt, rowvox);

    vfe_kernel<<<N_VOX, 256, 0, stream>>>(features, numvox,
                                          vfe1_W, vfe1_bn, vfe2_W, vfe2_bn,
                                          w3h, vfe3_bn, w4h, vfe4_bn,
                                          voxelwise);
    bfe12_kernel<<<(N_VOX + 3) / 4, 256, 0, stream>>>(voxelwise, rowvox, slotcnt, nrows,
                                                      bfe1_W, bfe1_bn, bfe2_W, bfe2_bn, x17);
    bfe3_kernel<<<(N_VOX + 31) / 32, 256, 0, stream>>>(x17, row_cell, nrows,
                                                       bfe3_W ? wb3h : wb3h, bfe3_bn, out);
}
